// MPNNet_Parametric_65970697667129
// MI455X (gfx1250) — matrix operand order repaired and checked by reading, not compiled, not run on hardware
//
#include <hip/hip_runtime.h>
#include <hip/hip_bf16.h>
#include <cstddef>

// ---------------- problem constants (match reference) ----------------
#define NN   25000
#define EE   100000
#define FEAT 16
#define DD   64
#define NG   512
#define JJ   1024
#define SS   2048
#define OO   105

#define MT   ((NN + 15) / 16)      // 1563 m-tiles
#define MTL  ((NN - 1) / 16)       // 1562 last valid m-tile
#define MG2  ((MT + 1) / 2)        // 782 blocks of 2 m-tiles

typedef _Float16 f16;
typedef __attribute__((ext_vector_type(16))) _Float16 v16h;
typedef __attribute__((ext_vector_type(8)))  _Float16 v8h;
typedef __attribute__((ext_vector_type(8)))  float    v8f;

__device__ __forceinline__ float leaky_f(float v){ return v > 0.f ? v : 0.01f * v; }
__device__ __forceinline__ float sigm_f(float v){ return 1.f / (1.f + expf(-v)); }

// order-preserving float<->uint encoding for atomic max
__device__ __forceinline__ unsigned fenc(float x){
    unsigned u = __float_as_uint(x);
    return (u & 0x80000000u) ? ~u : (u | 0x80000000u);
}
__device__ __forceinline__ float fdec(unsigned u){
    return (u & 0x80000000u) ? __uint_as_float(u ^ 0x80000000u) : __uint_as_float(~u);
}

// ---------------- WMMA fragment helpers (wave32) ----------------
// A: 16x32 f16 from row-major [.., lda]; contiguous k order, same as frag_bT/frag_bN:
// lane l: m = m0+(l&15), half = l>>4; VGPR v holds K pair at
// k = k0 + half*16 + v*2  (per-lane contiguous 32B chunk)
__device__ __forceinline__ v16h frag_a(const f16* __restrict__ A, int lda,
                                       int m0, int k0, int mmax){
    int l = threadIdx.x & 31;
    int m = m0 + (l & 15); if (m > mmax) m = mmax;
    int half = l >> 4;
    const f16* row = A + (size_t)m * lda;
    v16h a;
#pragma unroll
    for (int v = 0; v < 8; ++v){
        int kb = k0 + (half << 4) + (v << 1);
        a[2*v]   = row[kb];
        a[2*v+1] = row[kb+1];
    }
    return a;
}

// B: 32x16, B[k,n] = W[n*K + k]  (X @ W.T, W row-major [Nout,K])
__device__ __forceinline__ v16h frag_bT(const f16* __restrict__ W, int K,
                                        int n0, int k0){
    int l = threadIdx.x & 31;
    int n = n0 + (l & 15);
    int half = l >> 4;
    const f16* row = W + (size_t)n * K;
    v16h b;
#pragma unroll
    for (int v = 0; v < 8; ++v){
        int kb = k0 + (half << 4) + (v << 1);
        b[2*v]   = row[kb];
        b[2*v+1] = row[kb+1];
    }
    return b;
}

// B: 32x16, B[k,n] = W[k*ldn + n]  (X @ W, non-transposed weight)
__device__ __forceinline__ v16h frag_bN(const f16* __restrict__ W, int ldn,
                                        int n0, int k0){
    int l = threadIdx.x & 31;
    int n = n0 + (l & 15);
    int half = l >> 4;
    v16h b;
#pragma unroll
    for (int v = 0; v < 8; ++v){
        int kb = k0 + (half << 4) + (v << 1);
        b[2*v]   = W[(size_t)kb * ldn + n];
        b[2*v+1] = W[(size_t)(kb+1) * ldn + n];
    }
    return b;
}

__device__ __forceinline__ v8f wmma_f16(v16h a, v16h b, v8f c){
    return __builtin_amdgcn_wmma_f32_16x16x32_f16(false, a, false, b,
                                                  (short)0, c, false, false);
}

// ---------------- small utility kernels ----------------
__global__ void k_zero(float* p, long n){
    long i = (long)blockIdx.x * blockDim.x + threadIdx.x;
    if (i < n) p[i] = 0.f;
}

__global__ void k_cvt(const float* __restrict__ s, f16* __restrict__ d, int n){
    int i = blockIdx.x * blockDim.x + threadIdx.x;
    if (i < n) d[i] = (f16)s[i];
}

// Wp[c,i] = net2_w[(i*64 + (c>>6))*64 + (c&63)]   (c = o*64+k flattened)
__global__ void k_permw(const float* __restrict__ w2, f16* __restrict__ wp){
    int idx = blockIdx.x * blockDim.x + threadIdx.x;
    if (idx >= 4096 * DD) return;
    int c = idx >> 6, i = idx & 63;
    wp[idx] = (f16)w2[(size_t)(i * DD + (c >> 6)) * DD + (c & 63)];
}

// out = leaky(x @ lin0_w.T + b)  -> h (f32) and h (f16)
__global__ void k_lin0(const float* __restrict__ x, const float* __restrict__ w,
                       const float* __restrict__ b, float* __restrict__ hf,
                       f16* __restrict__ hh){
    int idx = blockIdx.x * blockDim.x + threadIdx.x;
    if (idx >= NN * DD) return;
    int n = idx >> 6, o = idx & 63;
    float acc = b[o];
#pragma unroll
    for (int k = 0; k < FEAT; ++k) acc += x[n*FEAT + k] * w[o*FEAT + k];
    acc = leaky_f(acc);
    hf[idx] = acc; hh[idx] = (f16)acc;
}

// e1 = leaky(edge_attr @ net1_w.T + b)  (kept f32: feeds VALU matvec)
__global__ void k_edge1(const float* __restrict__ ea, const float* __restrict__ w,
                        const float* __restrict__ b, float* __restrict__ e1){
    int idx = blockIdx.x * blockDim.x + threadIdx.x;
    if (idx >= EE * DD) return;
    int e = idx >> 6, o = idx & 63;
    float acc = b[o];
#pragma unroll
    for (int k = 0; k < 4; ++k) acc += ea[e*4 + k] * w[o*4 + k];
    e1[idx] = leaky_f(acc);
}

__global__ void k_deg(const int* __restrict__ dst, float* deg){
    int e = blockIdx.x * blockDim.x + threadIdx.x;
    if (e < EE) atomicAdd(&deg[dst[e]], 1.f);
}
__global__ void k_invdeg(const float* __restrict__ deg, float* inv){
    int n = blockIdx.x * blockDim.x + threadIdx.x;
    if (n < NN){ float d = deg[n]; inv[n] = d > 0.f ? 1.f / d : 0.f; }
}

// --------- T[n, o*64+k] = sum_i h[n,i] * Wp[o*64+k, i]   (WMMA GEMM) ------
// grid (MT, 32), block 256 = 8 waves; each wave one 16x16 tile, K=64.
// Epilogue transposes through LDS so global stores are contiguous b128.
__global__ void k_tmat(const f16* __restrict__ hh, const f16* __restrict__ wph,
                       f16* __restrict__ T){
    __shared__ f16 sm[16 * 128];
    int wave = threadIdx.x >> 5;
    int m0 = blockIdx.x * 16;
    int nb0 = blockIdx.y * 128;
    int n0 = nb0 + wave * 16;
    v16h a0 = frag_a(hh, DD, m0, 0,  NN - 1);
    v16h a1 = frag_a(hh, DD, m0, 32, NN - 1);
    v8f c = {};
    c = wmma_f16(a0, frag_bT(wph, DD, n0, 0),  c);
    c = wmma_f16(a1, frag_bT(wph, DD, n0, 32), c);
    int l = threadIdx.x & 31, half = l >> 4;
    int lc = wave * 16 + (l & 15);
#pragma unroll
    for (int r = 0; r < 8; ++r) sm[(r + half * 8) * 128 + lc] = (f16)c[r];
    __syncthreads();
    int row = threadIdx.x >> 4;          // 0..15
    int ch  = threadIdx.x & 15;          // 16B chunk within 256B row
    int m = m0 + row;
    if (m < NN){
        v8h val = *(const v8h*)(sm + row * 128 + ch * 8);
        *(v8h*)(T + (size_t)m * 4096 + nb0 + ch * 8) = val;
    }
}

// --------- Bias[n,o] = sum_i h[n,i] * net2_b[i*64+o]   (small WMMA) -------
__global__ void k_bias(const f16* __restrict__ hh, const f16* __restrict__ nbh,
                       float* __restrict__ biasn){
    int wave = threadIdx.x >> 5;
    int mt = blockIdx.x * 2 + (wave >> 2); if (mt > MTL) mt = MTL;
    int m0 = mt * 16;
    int n0 = (wave & 3) * 16;
    v16h a0 = frag_a(hh, DD, m0, 0,  NN - 1);
    v16h a1 = frag_a(hh, DD, m0, 32, NN - 1);
    v8f c = {};
    c = wmma_f16(a0, frag_bN(nbh, DD, n0, 0),  c);
    c = wmma_f16(a1, frag_bN(nbh, DD, n0, 32), c);
    int l = threadIdx.x & 31;
    int n = n0 + (l & 15), half = l >> 4;
#pragma unroll
    for (int r = 0; r < 8; ++r){
        int m = m0 + r + half * 8;
        if (m < NN) biasn[(size_t)m * DD + n] = c[r];
    }
}

// --------- per-edge msg[o] = dot(e1[e,:], T[src,o,:]) + Bias[src,o] -------
// one wave per edge, 2 edges per wave with prefetch of the 2nd gather row
__device__ __forceinline__ void msg_one(const float* __restrict__ ev,
                                        const f16* __restrict__ trow,
                                        const float* __restrict__ bp,
                                        float* __restrict__ ap){
    const v8h* tA = (const v8h*)trow;          // o = 2*lane   row (64 f16)
    float a0 = 0.f, a1 = 0.f;
#pragma unroll
    for (int kk = 0; kk < 8; ++kk){
        v8h wa = tA[kk];
        v8h wb = tA[kk + 8];                   // o = 2*lane+1 row
#pragma unroll
        for (int j = 0; j < 8; ++j){
            float cv = ev[kk * 8 + j];
            a0 += cv * (float)wa[j];
            a1 += cv * (float)wb[j];
        }
    }
    atomicAdd(ap,     a0 + bp[0]);
    atomicAdd(ap + 1, a1 + bp[1]);
}

__global__ void k_msg(const float* __restrict__ e1, const f16* __restrict__ T,
                      const float* __restrict__ biasn,
                      const int* __restrict__ src, const int* __restrict__ dst,
                      float* __restrict__ aggr){
    int wave = threadIdx.x >> 5, lane = threadIdx.x & 31;
    int e = blockIdx.x * 16 + wave * 2;          // 2 edges per wave
    if (e + 1 >= EE + 1) return;                 // EE divisible by 16; keep guard trivial
    int s0 = src[e],     d0 = dst[e];
    int s1 = src[e + 1], d1 = dst[e + 1];
    const f16* r0 = T + (size_t)s0 * 4096 + lane * 128;
    const f16* r1 = T + (size_t)s1 * 4096 + lane * 128;
    __builtin_prefetch(r1, 0, 0);                // overlap gather latency
    __builtin_prefetch(r1 + 64, 0, 0);
    msg_one(e1 + (size_t)e * DD,       r0,
            biasn + (size_t)s0 * DD + lane * 2,
            aggr  + (size_t)d0 * DD + lane * 2);
    msg_one(e1 + (size_t)(e + 1) * DD, r1,
            biasn + (size_t)s1 * DD + lane * 2,
            aggr  + (size_t)d1 * DD + lane * 2);
}

// --------- fused NNConv-root + GRU cell (conv result exchanged via LDS) ---
// block 256 = 8 waves: 2 m-tiles x 4 col-slices; no early returns (barrier!)
__global__ void k_convgru(const f16* __restrict__ hh, const float* __restrict__ hf,
                          const f16* __restrict__ crh, const float* __restrict__ aggr,
                          const float* __restrict__ invd, const float* __restrict__ cb,
                          const f16* __restrict__ wih, const f16* __restrict__ whh,
                          const float* __restrict__ bih, const float* __restrict__ bhh,
                          float* __restrict__ hf_o, f16* __restrict__ hh_o){
    __shared__ f16 sm[2][16 * DD];
    int wave = threadIdx.x >> 5;
    int mtl = wave >> 2;
    int mt = blockIdx.x * 2 + mtl; if (mt > MTL) mt = MTL;
    int m0 = mt * 16;
    int n0 = (wave & 3) * 16;
    int l = threadIdx.x & 31;
    int n = n0 + (l & 15), half = l >> 4;

    // h fragments (used for conv A and GRU gh A)
    v16h a0 = frag_a(hh, DD, m0, 0,  NN - 1);
    v16h a1 = frag_a(hh, DD, m0, 32, NN - 1);

    // phase 1: m = leaky(aggr*invdeg + h @ conv_root + conv_b) -> LDS
    v8f c = {};
    c = wmma_f16(a0, frag_bN(crh, DD, n0, 0),  c);
    c = wmma_f16(a1, frag_bN(crh, DD, n0, 32), c);
    float bv = cb[n];
#pragma unroll
    for (int r = 0; r < 8; ++r){
        int mr = r + half * 8;
        int m = m0 + mr; int mc = m < NN ? m : NN - 1;
        float v = c[r] + aggr[(size_t)mc * DD + n] * invd[mc] + bv;
        sm[mtl][mr * DD + n] = (f16)leaky_f(v);
    }
    __syncthreads();

    // phase 2: GRU gates. gi from m (LDS), gh from h (a0/a1 reused)
    const f16* mrow = &sm[mtl][0];
    v16h p0 = frag_a(mrow, DD, 0, 0,  15);
    v16h p1 = frag_a(mrow, DD, 0, 32, 15);
    v8f cRi = {}, cZi = {}, cNi = {}, cRh = {}, cZh = {}, cNh = {};
    cRi = wmma_f16(p0, frag_bT(wih, DD,       n0, 0),  cRi);
    cRi = wmma_f16(p1, frag_bT(wih, DD,       n0, 32), cRi);
    cZi = wmma_f16(p0, frag_bT(wih, DD,  64 + n0, 0),  cZi);
    cZi = wmma_f16(p1, frag_bT(wih, DD,  64 + n0, 32), cZi);
    cNi = wmma_f16(p0, frag_bT(wih, DD, 128 + n0, 0),  cNi);
    cNi = wmma_f16(p1, frag_bT(wih, DD, 128 + n0, 32), cNi);
    cRh = wmma_f16(a0, frag_bT(whh, DD,       n0, 0),  cRh);
    cRh = wmma_f16(a1, frag_bT(whh, DD,       n0, 32), cRh);
    cZh = wmma_f16(a0, frag_bT(whh, DD,  64 + n0, 0),  cZh);
    cZh = wmma_f16(a1, frag_bT(whh, DD,  64 + n0, 32), cZh);
    cNh = wmma_f16(a0, frag_bT(whh, DD, 128 + n0, 0),  cNh);
    cNh = wmma_f16(a1, frag_bT(whh, DD, 128 + n0, 32), cNh);

    float bRi = bih[n],       bRh = bhh[n];
    float bZi = bih[64 + n],  bZh = bhh[64 + n];
    float bNi = bih[128 + n], bNh = bhh[128 + n];
#pragma unroll
    for (int r = 0; r < 8; ++r){
        int m = m0 + r + half * 8;
        if (m < NN){
            float rg = sigm_f(cRi[r] + bRi + cRh[r] + bRh);
            float zg = sigm_f(cZi[r] + bZi + cZh[r] + bZh);
            float ng = tanhf(cNi[r] + bNi + rg * (cNh[r] + bNh));
            float hp = hf[(size_t)m * DD + n];
            float hn = (1.f - zg) * ng + zg * hp;
            hf_o[(size_t)m * DD + n] = hn;
            hh_o[(size_t)m * DD + n] = (f16)hn;
        }
    }
}

// ---------------- Set2Set (1 step, zero init -> q shared by all graphs) ----
__global__ void k_qv(const float* __restrict__ bih, const float* __restrict__ bhh,
                     float* __restrict__ qv){
    int t = threadIdx.x;
    if (t < DD){
        float gi = bih[t]        + bhh[t];
        float gg = bih[128 + t]  + bhh[128 + t];
        float go = bih[192 + t]  + bhh[192 + t];
        float c  = sigm_f(gi) * tanhf(gg);
        qv[t] = sigm_f(go) * tanhf(c);
    }
}

__global__ void k_attn1(const float* __restrict__ hf, const float* __restrict__ qv,
                        const int* __restrict__ batch, float* __restrict__ en,
                        unsigned* __restrict__ emax){
    int n = blockIdx.x * blockDim.x + threadIdx.x;
    if (n >= NN) return;
    float e = 0.f;
#pragma unroll 8
    for (int o = 0; o < DD; ++o) e += hf[(size_t)n*DD + o] * qv[o];
    en[n] = e;
    atomicMax(&emax[batch[n]], fenc(e));
}

__global__ void k_attn2(const float* __restrict__ en, const int* __restrict__ batch,
                        const unsigned* __restrict__ emax, float* __restrict__ exn,
                        float* __restrict__ denom){
    int n = blockIdx.x * blockDim.x + threadIdx.x;
    if (n >= NN) return;
    int b = batch[n];
    float ex = expf(en[n] - fdec(emax[b]));
    exn[n] = ex;
    atomicAdd(&denom[b], ex);
}

__global__ void k_attn3(const float* __restrict__ hf, const float* __restrict__ exn,
                        const float* __restrict__ denom, const int* __restrict__ batch,
                        float* __restrict__ rvec){
    int idx = blockIdx.x * blockDim.x + threadIdx.x;
    if (idx >= NN * DD) return;
    int n = idx >> 6, o = idx & 63;
    int b = batch[n];
    float a = exn[n] / denom[b];
    atomicAdd(&rvec[(size_t)b*DD + o], a * hf[idx]);
}

__global__ void k_gout(const float* __restrict__ qv, const float* __restrict__ rvec,
                       const float* __restrict__ lw, const float* __restrict__ lb,
                       float* __restrict__ gout){
    int idx = blockIdx.x * blockDim.x + threadIdx.x;
    if (idx >= NG * 2) return;
    int g = idx >> 1, oi = idx & 1;
    float acc = lb[oi];
#pragma unroll 8
    for (int k = 0; k < DD; ++k) acc += qv[k] * lw[oi*128 + k];
#pragma unroll 8
    for (int k = 0; k < DD; ++k) acc += rvec[(size_t)g*DD + k] * lw[oi*128 + 64 + k];
    gout[idx] = acc;
}

// ---------------- stem head: leaky(h[s]@W1.T+b1)@W2.T+b2 -------------------
__global__ void k_stem(const float* __restrict__ hf, const int* __restrict__ sidx,
                       const float* __restrict__ w1, const float* __restrict__ b1,
                       const float* __restrict__ w2, const float* __restrict__ b2,
                       float* __restrict__ outp){
    __shared__ float hid[DD];
    int s = blockIdx.x, t = threadIdx.x;
    int row = sidx[s];
    if (t < DD){
        float a = b1[t];
#pragma unroll 8
        for (int k = 0; k < DD; ++k) a += hf[(size_t)row*DD + k] * w1[t*DD + k];
        hid[t] = leaky_f(a);
    }
    __syncthreads();
    for (int c = t; c < OO; c += blockDim.x){
        float a = b2[c];
#pragma unroll 8
        for (int k = 0; k < DD; ++k) a += hid[k] * w2[c*DD + k];
        outp[(size_t)s*OO + c] = a;
    }
}

// ---------------- jbond head: mean over pair of scalar MLP outputs ---------
__global__ void k_jbond(const float* __restrict__ hf, const int* __restrict__ jidx,
                        const float* __restrict__ w1, const float* __restrict__ b1,
                        const float* __restrict__ w2, const float* __restrict__ b2,
                        float* __restrict__ outp){
    __shared__ float red[DD];
    int p = blockIdx.x, t = threadIdx.x;
    int r0 = jidx[p*2 + 0], r1 = jidx[p*2 + 1];
    float h0 = b1[t], h1 = b1[t];
#pragma unroll 8
    for (int k = 0; k < DD; ++k){
        float wv = w1[t*DD + k];
        h0 += hf[(size_t)r0*DD + k] * wv;
        h1 += hf[(size_t)r1*DD + k] * wv;
    }
    red[t] = (leaky_f(h0) + leaky_f(h1)) * w2[t];
    __syncthreads();
    for (int s2 = 32; s2 > 0; s2 >>= 1){
        if (t < s2) red[t] += red[t + s2];
        __syncthreads();
    }
    if (t == 0) outp[p] = 0.5f * red[0] + b2[0];
}

// ======================= host side =========================================
extern "C" void kernel_launch(void* const* d_in, const int* in_sizes, int n_in,
                              void* d_out, int out_size, void* d_ws, size_t ws_size,
                              hipStream_t stream){
    const float* x        = (const float*)d_in[0];
    const int*   eidx     = (const int*)  d_in[1];
    const float* eattr    = (const float*)d_in[2];
    const int*   jidx     = (const int*)  d_in[3];
    const int*   sidx     = (const int*)  d_in[4];
    const int*   batch    = (const int*)  d_in[5];
    const float* lin0_w   = (const float*)d_in[6];
    const float* lin0_b   = (const float*)d_in[7];
    const float* net1_w   = (const float*)d_in[8];
    const float* net1_b   = (const float*)d_in[9];
    const float* net2_w   = (const float*)d_in[10];
    const float* net2_b   = (const float*)d_in[11];
    const float* conv_root= (const float*)d_in[12];
    const float* conv_b   = (const float*)d_in[13];
    const float* gru_wih  = (const float*)d_in[14];
    const float* gru_whh  = (const float*)d_in[15];
    const float* gru_bih  = (const float*)d_in[16];
    const float* gru_bhh  = (const float*)d_in[17];
    const float* n2s_w1   = (const float*)d_in[18];
    const float* n2s_b1   = (const float*)d_in[19];
    const float* n2s_w2   = (const float*)d_in[20];
    const float* n2s_b2   = (const float*)d_in[21];
    const float* n2j_w1   = (const float*)d_in[22];
    const float* n2j_b1   = (const float*)d_in[23];
    const float* n2j_w2   = (const float*)d_in[24];
    const float* n2j_b2   = (const float*)d_in[25];
    const float* lstm_bih = (const float*)d_in[28];
    const float* lstm_bhh = (const float*)d_in[29];
    const float* lout_w   = (const float*)d_in[30];
    const float* lout_b   = (const float*)d_in[31];
    float* outp = (float*)d_out;

    // bump allocator over workspace
    char* base = (char*)d_ws;
    size_t off = 0;
    auto alloc = [&](size_t bytes) -> void* {
        void* p = base + off;
        off += (bytes + 255) & ~(size_t)255;
        return p;
    };
    float* hA      = (float*)alloc((size_t)NN * DD * 4);
    float* hB      = (float*)alloc((size_t)NN * DD * 4);
    f16*   hhA     = (f16*)  alloc((size_t)NN * DD * 2);
    f16*   hhB     = (f16*)  alloc((size_t)NN * DD * 2);
    float* aggr    = (float*)alloc((size_t)NN * DD * 4);
    float* biasn   = (float*)alloc((size_t)NN * DD * 4);
    float* deg     = (float*)alloc((size_t)NN * 4);
    float* invd    = (float*)alloc((size_t)NN * 4);
    float* e1      = (float*)alloc((size_t)EE * DD * 4);
    f16*   wph     = (f16*)  alloc((size_t)4096 * DD * 2);
    f16*   nbh     = (f16*)  alloc((size_t)4096 * 2);
    f16*   crh     = (f16*)  alloc((size_t)DD * DD * 2);
    f16*   wihh    = (f16*)  alloc((size_t)192 * DD * 2);
    f16*   whhh    = (f16*)  alloc((size_t)192 * DD * 2);
    float* qv      = (float*)alloc((size_t)DD * 4);
    float* en      = (float*)alloc((size_t)NN * 4);
    float* exn     = (float*)alloc((size_t)NN * 4);
    unsigned* emax = (unsigned*)alloc((size_t)NG * 4);
    float* denom   = (float*)alloc((size_t)NG * 4);
    float* rvec    = (float*)alloc((size_t)NG * DD * 4);
    f16*   T       = (f16*)  alloc((size_t)NN * 4096 * 2);   // 205 MB, ~L2-resident

    const int* src = eidx;
    const int* dst = eidx + EE;

    const int B = 256;
    auto cdiv = [](long a, long b){ return (int)((a + b - 1) / b); };

    // ---- one-time precompute ----
    k_zero<<<cdiv(NN,B), B, 0, stream>>>(deg, NN);
    k_lin0 <<<cdiv((long)NN*DD,B), B, 0, stream>>>(x, lin0_w, lin0_b, hA, hhA);
    k_edge1<<<cdiv((long)EE*DD,B), B, 0, stream>>>(eattr, net1_w, net1_b, e1);
    k_deg  <<<cdiv(EE,B), B, 0, stream>>>(dst, deg);
    k_invdeg<<<cdiv(NN,B), B, 0, stream>>>(deg, invd);
    k_permw<<<cdiv(4096*DD,B), B, 0, stream>>>(net2_w, wph);
    k_cvt<<<cdiv(4096,B),   B, 0, stream>>>(net2_b,   nbh,  4096);
    k_cvt<<<cdiv(DD*DD,B),  B, 0, stream>>>(conv_root,crh,  DD*DD);
    k_cvt<<<cdiv(192*DD,B), B, 0, stream>>>(gru_wih,  wihh, 192*DD);
    k_cvt<<<cdiv(192*DD,B), B, 0, stream>>>(gru_whh,  whhh, 192*DD);

    // ---- 6 message-passing iterations (ping-pong h buffers) ----
    float* hf = hA;  f16* hh = hhA;
    float* hg = hB;  f16* hg2 = hhB;
    for (int it = 0; it < 6; ++it){
        k_zero<<<cdiv((long)NN*DD,B), B, 0, stream>>>(aggr, (long)NN*DD);
        k_tmat<<<dim3(MT, 32), B, 0, stream>>>(hh, wph, T);
        k_bias<<<MG2, B, 0, stream>>>(hh, nbh, biasn);
        k_msg <<<EE/16, B, 0, stream>>>(e1, T, biasn, src, dst, aggr);
        k_convgru<<<MG2, B, 0, stream>>>(hh, hf, crh, aggr, invd, conv_b,
                                         wihh, whhh, gru_bih, gru_bhh, hg, hg2);
        float* tf = hf; hf = hg; hg = tf;
        f16*   th = hh; hh = hg2; hg2 = th;
    }

    // ---- Set2Set attention readout ----
    k_qv<<<1, 64, 0, stream>>>(lstm_bih, lstm_bhh, qv);
    k_zero<<<1, B, 0, stream>>>((float*)emax, NG);           // 0u encodes "-inf"
    k_zero<<<1, B, 0, stream>>>(denom, NG);
    k_zero<<<cdiv((long)NG*DD,B), B, 0, stream>>>(rvec, (long)NG*DD);
    k_attn1<<<cdiv(NN,B), B, 0, stream>>>(hf, qv, batch, en, emax);
    k_attn2<<<cdiv(NN,B), B, 0, stream>>>(en, batch, emax, exn, denom);
    k_attn3<<<cdiv((long)NN*DD,B), B, 0, stream>>>(hf, exn, denom, batch, rvec);
    k_gout <<<cdiv(NG*2,B), B, 0, stream>>>(qv, rvec, lout_w, lout_b, outp);

    // ---- heads (output layout: graph_out | stem | jbond) ----
    float* stem_out  = outp + NG * 2;
    float* jbond_out = stem_out + (size_t)SS * OO;
    k_stem <<<SS, 128, 0, stream>>>(hf, sidx, n2s_w1, n2s_b1, n2s_w2, n2s_b2, stem_out);
    k_jbond<<<JJ, 64,  0, stream>>>(hf, jidx, n2j_w1, n2j_b1, n2j_w2, n2j_b2, jbond_out);

    (void)in_sizes; (void)n_in; (void)out_size; (void)ws_size;
}
